// GroupQueryAttention_11768210391577
// MI455X (gfx1250) — matrix operand order repaired and checked by reading, not compiled, not run on hardware
//
#include <hip/hip_runtime.h>
#include <hip/hip_bf16.h>
#include <cstdint>

// ---------------------------------------------------------------------------
// GQA attention for MI455X (gfx1250): all matmuls via v_wmma_f32_16x16x32_bf16,
// GEMM A-tiles staged with GLOBAL_LOAD_ASYNC_TO_LDS_B128 (ASYNCcnt pipeline),
// double-buffered LDS.
// ---------------------------------------------------------------------------

typedef __attribute__((ext_vector_type(16))) __bf16 v16bf;
typedef __attribute__((ext_vector_type(8)))  __bf16 bf16x8;
typedef __attribute__((ext_vector_type(4)))  __bf16 bf16x4;
typedef __attribute__((ext_vector_type(2)))  __bf16 bf16x2;
typedef __attribute__((ext_vector_type(8)))  float  v8f;

constexpr int S_LEN = 2048;
constexpr int DIN   = 4096;
constexpr int NH    = 32;
constexpr int NKV   = 8;
constexpr int DH    = 128;
constexpr int GQ    = NH / NKV;   // 4

// ---------------------------------------------------------------------------
// fp32 -> bf16 bulk convert (X pre-pass). 4 elements / thread, float4 reads.
// ---------------------------------------------------------------------------
__global__ __launch_bounds__(256)
void cvt_f32_bf16(const float* __restrict__ in, __bf16* __restrict__ out)
{
  int i = (blockIdx.x * blockDim.x + threadIdx.x) * 4;
  float4 v = *(const float4*)(in + i);
  bf16x4 o;
  o[0] = (__bf16)v.x; o[1] = (__bf16)v.y; o[2] = (__bf16)v.z; o[3] = (__bf16)v.w;
  *(bf16x4*)(out + i) = o;
}

// ---------------------------------------------------------------------------
// Tiled GEMM: C[M,N] = A[M,K](bf16) * B[K,N](fp32, converted to bf16 in LDS).
// Block tile 128x128, K step 32, 256 threads (8 waves as 2x4; each wave owns a
// 64x32 tile = 8 WMMA accumulators). Double-buffered LDS:
//   - A tile staged via global_load_async_to_lds_b128 (no VGPR roundtrip),
//     pipelined with s_wait_asynccnt (in-order ASYNCcnt: wait<=2 means the
//     previous tile's 2 asyncs have retired).
//   - B tile converted fp32->bf16 and stored transposed ([col][k], stride 40
//     -> 80B lane stride, ~2-way banking) with packed 4B stores.
// ---------------------------------------------------------------------------
template <typename OutT>
__global__ __launch_bounds__(256)
void gemm_bf16wmma(const __bf16* __restrict__ A, const float* __restrict__ B,
                   OutT* __restrict__ C, int M, int N, int Kd)
{
  __shared__ __bf16 As[2][128 * 40];   // [row][k] stride 40
  __shared__ __bf16 Bs[2][128 * 40];   // [col][k] stride 40 (transposed)

  const int tid  = threadIdx.x;
  const int lane = tid & 31;
  const int wave = tid >> 5;
  const int ln   = lane & 15;
  const int hi   = lane >> 4;
  const int wm   = wave >> 2;           // 0..1
  const int wn   = wave & 3;            // 0..3
  const int m0   = blockIdx.y * 128;
  const int n0   = blockIdx.x * 128;

  v8f acc[4][2];
  #pragma unroll
  for (int a = 0; a < 4; ++a)
    #pragma unroll
    for (int b = 0; b < 2; ++b)
      #pragma unroll
      for (int j = 0; j < 8; ++j) acc[a][b][j] = 0.0f;

  const int bc  = tid & 127;            // B: column handled by this thread
  const int bkh = (tid >> 7) * 16;      // B: first k-row of this thread's half

  const int nk = Kd / 32;

  auto stage = [&](int buf, int k) {
    // ---- A tile (128 x 32 bf16 = 8KB): 2 async b128 copies per thread ----
    #pragma unroll
    for (int i = 0; i < 2; ++i) {
      int idx = i * 256 + tid;
      int row = idx >> 2;
      int kc  = (idx & 3) * 8;
      unsigned ldsa = (unsigned)(size_t)(&As[buf][row * 40 + kc]);
      uint64_t ga   = (uint64_t)(A + (size_t)(m0 + row) * Kd + k + kc);
      asm volatile("global_load_async_to_lds_b128 %0, %1, off"
                   :: "v"(ldsa), "v"(ga) : "memory");
    }
    // ---- B tile (32 x 128 fp32): coalesced row reads, transposed stores ----
    const float* bp = B + (size_t)(k + bkh) * N + n0 + bc;
    #pragma unroll
    for (int i = 0; i < 16; i += 2) {
      float f0 = bp[(size_t)i * N];
      float f1 = bp[(size_t)(i + 1) * N];
      bf16x2 p;
      p[0] = (__bf16)f0; p[1] = (__bf16)f1;
      *(bf16x2*)(&Bs[buf][bc * 40 + bkh + i]) = p;
    }
  };

  stage(0, 0);
  for (int t = 0; t < nk; ++t) {
    const int cur = t & 1;
    __syncthreads();                       // prev compute done before overwrite
    if (t + 1 < nk) {
      stage(1 - cur, (t + 1) * 32);
      asm volatile("s_wait_asynccnt 0x2" ::: "memory");   // cur A-tile landed
    } else {
      asm volatile("s_wait_asynccnt 0x0" ::: "memory");
    }
    __syncthreads();                       // cur tile visible to all waves

    // A fragments: lane ln = row; halves h: K = hi*16 + h
    v16bf af[4];
    #pragma unroll
    for (int ms = 0; ms < 4; ++ms) {
      const __bf16* ap = &As[cur][(wm * 64 + ms * 16 + ln) * 40];
      #pragma unroll
      for (int g = 0; g < 2; ++g) {
        bf16x8 v = *(const bf16x8*)(ap + hi * 16 + g * 8);
        #pragma unroll
        for (int e = 0; e < 8; ++e) af[ms][g * 8 + e] = v[e];
      }
    }
    // B fragments: lane ln = col; halves h: K = hi*16 + h
    v16bf bfr[2];
    #pragma unroll
    for (int ns = 0; ns < 2; ++ns) {
      const __bf16* bp2 = &Bs[cur][(wn * 32 + ns * 16 + ln) * 40 + hi * 16];
      bf16x8 v0 = *(const bf16x8*)(bp2);
      bf16x8 v1 = *(const bf16x8*)(bp2 + 8);
      #pragma unroll
      for (int e = 0; e < 8; ++e) { bfr[ns][e] = v0[e]; bfr[ns][8 + e] = v1[e]; }
    }

    #pragma unroll
    for (int ms = 0; ms < 4; ++ms)
      #pragma unroll
      for (int ns = 0; ns < 2; ++ns)
        acc[ms][ns] = __builtin_amdgcn_wmma_f32_16x16x32_bf16(
            false, af[ms], false, bfr[ns], (short)0, acc[ms][ns], false, false);
  }

  // C/D layout: element j -> row = j + hi*8, col = ln
  #pragma unroll
  for (int ms = 0; ms < 4; ++ms)
    #pragma unroll
    for (int ns = 0; ns < 2; ++ns)
      #pragma unroll
      for (int j = 0; j < 8; ++j) {
        int row = m0 + wm * 64 + ms * 16 + j + hi * 8;
        int col = n0 + wn * 32 + ns * 16 + ln;
        C[(size_t)row * N + col] = (OutT)acc[ms][ns][j];
      }
}

// ---------------------------------------------------------------------------
// RoPE (rotate-half) in place on bf16 [S, heads, 128]; optional scale
// (1/sqrt(DH) folded into Q so attention needs no extra score scaling).
// ---------------------------------------------------------------------------
__global__ __launch_bounds__(256)
void rope_kernel(__bf16* __restrict__ t, const float* __restrict__ cosT,
                 const float* __restrict__ sinT, int heads, float scale)
{
  int idx = blockIdx.x * blockDim.x + threadIdx.x;   // s*heads*64 threads
  int d   = idx & 63;
  int hd  = (idx >> 6) % heads;
  int s   = idx / (heads * 64);
  __bf16* p = t + (size_t)s * heads * DH + hd * DH;
  float t1 = (float)p[d];
  float t2 = (float)p[d + 64];
  float c1 = cosT[s * DH + d],      s1 = sinT[s * DH + d];
  float c2 = cosT[s * DH + d + 64], s2 = sinT[s * DH + d + 64];
  p[d]      = (__bf16)((t1 * c1 - t2 * s1) * scale);
  p[d + 64] = (__bf16)((t2 * c2 + t1 * s2) * scale);
}

// ---------------------------------------------------------------------------
// Flash attention (causal, GQA). Block = 1 head x 64 q rows, 4 waves x 16 rows.
// Per 32-key chunk: block stages V^T into LDS; scores = 8 WMMAs (K fragments
// contiguous from row-major K); online softmax in C-layout with shfl_xor row
// reductions; P re-laid out C->A via a per-wave padded LDS tile; ctx += P*V
// with 8 WMMAs off the V^T tile. Block-uniform causal loop bound keeps
// __syncthreads legal; above-diagonal chunks contribute exact zeros.
// Output written directly as bf16 so the output projection GEMM can async-
// stage it as its A operand.
// ---------------------------------------------------------------------------
__global__ __launch_bounds__(128)
void gqa_flash_attn(const __bf16* __restrict__ Q, const __bf16* __restrict__ K,
                    const __bf16* __restrict__ V, __bf16* __restrict__ O)
{
  __shared__ __bf16 vt[DH * 40];        // V^T chunk: [dh][key], stride 40
  __shared__ __bf16 pb[4 * 16 * 40];    // per-wave P tile: [row][key], stride 40

  const int h    = blockIdx.y;
  const int qt   = blockIdx.x;
  const int kvh  = h / GQ;
  const int tid  = threadIdx.x;
  const int lane = tid & 31;
  const int wave = tid >> 5;
  const int ln   = lane & 15;
  const int hi   = lane >> 4;
  const int q0   = qt * 64 + wave * 16;

  // preload Q A-fragments: 4 chunks of K=32 across DH=128
  v16bf qf[4];
  {
    const __bf16* qp = Q + (size_t)(q0 + ln) * (NH * DH) + h * DH;
    #pragma unroll
    for (int f = 0; f < 4; ++f)
      #pragma unroll
      for (int g = 0; g < 2; ++g) {
        bf16x8 v = *(const bf16x8*)(qp + f * 32 + hi * 16 + g * 8);
        #pragma unroll
        for (int e = 0; e < 8; ++e) qf[f][g * 8 + e] = v[e];
      }
  }

  float mrow[8], lrow[8];
  v8f acc[8];
  #pragma unroll
  for (int j = 0; j < 8; ++j) { mrow[j] = -__builtin_inff(); lrow[j] = 0.0f; }
  #pragma unroll
  for (int n = 0; n < 8; ++n)
    #pragma unroll
    for (int j = 0; j < 8; ++j) acc[n][j] = 0.0f;

  const int kend = qt * 64 + 64;        // uniform causal bound for the block
  const int vkey = tid >> 4;            // 0..7
  const int vdh  = (tid & 15) * 8;

  for (int kb = 0; kb < kend; kb += 32) {
    __syncthreads();
    // cooperatively stage V^T chunk (32 keys x 128 dh)
    #pragma unroll
    for (int i = 0; i < 4; ++i) {
      int key = i * 8 + vkey;
      bf16x8 vv = *(const bf16x8*)(V + (size_t)(kb + key) * (NKV * DH) + kvh * DH + vdh);
      #pragma unroll
      for (int e = 0; e < 8; ++e) vt[(vdh + e) * 40 + key] = vv[e];
    }
    __syncthreads();

    // scores for two 16-key subtiles
    v8f s0, s1;
    #pragma unroll
    for (int j = 0; j < 8; ++j) { s0[j] = 0.0f; s1[j] = 0.0f; }
    #pragma unroll
    for (int f = 0; f < 4; ++f) {
      // B fragment of K^T: lane ln = key column; halves h -> dh = f*32 + hi*16 + h
      v16bf k0, k1;
      const __bf16* kp  = K + (size_t)(kb + ln) * (NKV * DH) + kvh * DH + f * 32 + hi * 16;
      const __bf16* kp2 = kp + (size_t)16 * (NKV * DH);
      bf16x8 a0 = *(const bf16x8*)(kp);
      bf16x8 a1 = *(const bf16x8*)(kp + 8);
      bf16x8 b0 = *(const bf16x8*)(kp2);
      bf16x8 b1 = *(const bf16x8*)(kp2 + 8);
      #pragma unroll
      for (int e = 0; e < 8; ++e) {
        k0[e] = a0[e]; k0[8 + e] = a1[e];
        k1[e] = b0[e]; k1[8 + e] = b1[e];
      }
      s0 = __builtin_amdgcn_wmma_f32_16x16x32_bf16(false, qf[f], false, k0, (short)0, s0, false, false);
      s1 = __builtin_amdgcn_wmma_f32_16x16x32_bf16(false, qf[f], false, k1, (short)0, s1, false, false);
    }

    // online softmax update (element j -> row j + hi*8, col = ln)
    __bf16* pw = pb + (wave * 16) * 40;
    #pragma unroll
    for (int j = 0; j < 8; ++j) {
      const int row = q0 + j + hi * 8;
      float x0 = (kb + ln      <= row) ? s0[j] : -__builtin_inff();
      float x1 = (kb + 16 + ln <= row) ? s1[j] : -__builtin_inff();
      float mx = fmaxf(x0, x1);
      mx = fmaxf(mx, __shfl_xor(mx, 1));
      mx = fmaxf(mx, __shfl_xor(mx, 2));
      mx = fmaxf(mx, __shfl_xor(mx, 4));
      mx = fmaxf(mx, __shfl_xor(mx, 8));
      const float mnew  = fmaxf(mrow[j], mx);
      const float alpha = __expf(mrow[j] - mnew);
      const float p0 = __expf(x0 - mnew);
      const float p1 = __expf(x1 - mnew);
      float sum = p0 + p1;
      sum += __shfl_xor(sum, 1);
      sum += __shfl_xor(sum, 2);
      sum += __shfl_xor(sum, 4);
      sum += __shfl_xor(sum, 8);
      lrow[j] = lrow[j] * alpha + sum;
      mrow[j] = mnew;
      #pragma unroll
      for (int n = 0; n < 8; ++n) acc[n][j] *= alpha;
      __bf16* prow = pw + (j + hi * 8) * 40;
      prow[ln]      = (__bf16)p0;
      prow[ln + 16] = (__bf16)p1;
    }
    __syncthreads();   // orders P LDS writes vs cross-lane fragment reads

    // P (16x32) A-fragment from LDS
    v16bf pf;
    {
      const __bf16* pp = pb + (wave * 16 + ln) * 40;
      #pragma unroll
      for (int g = 0; g < 2; ++g) {
        bf16x8 v = *(const bf16x8*)(pp + hi * 16 + g * 8);
        #pragma unroll
        for (int e = 0; e < 8; ++e) pf[g * 8 + e] = v[e];
      }
    }
    // ctx += P * V  (B fragments from V^T tile: lane ln = dh col, K = key)
    #pragma unroll
    for (int n = 0; n < 8; ++n) {
      v16bf vf;
      const __bf16* vp = vt + (n * 16 + ln) * 40 + hi * 16;
      bf16x8 v0 = *(const bf16x8*)(vp);
      bf16x8 v1 = *(const bf16x8*)(vp + 8);
      #pragma unroll
      for (int e = 0; e < 8; ++e) { vf[e] = v0[e]; vf[8 + e] = v1[e]; }
      acc[n] = __builtin_amdgcn_wmma_f32_16x16x32_bf16(false, pf, false, vf, (short)0, acc[n], false, false);
    }
  }

  // epilogue: divide by softmax denominator, write bf16 context
  #pragma unroll
  for (int j = 0; j < 8; ++j) {
    const int row = q0 + j + hi * 8;
    const float inv = 1.0f / lrow[j];
    #pragma unroll
    for (int n = 0; n < 8; ++n)
      O[(size_t)row * (NH * DH) + h * DH + n * 16 + ln] = (__bf16)(acc[n][j] * inv);
  }
}

// ---------------------------------------------------------------------------
// Host-side launch
// inputs: 0=input_tensor f32, 1=attention_mask (unused; causal analytic),
//         2=cos f32, 3=sin f32, 4=Wq, 5=Wk, 6=Wv, 7=Wo
// workspace: Xbf 16MB | Qbf 16MB | Kbf 4MB | Vbf 4MB | Ctxbf 16MB (= 56MB)
// ---------------------------------------------------------------------------
extern "C" void kernel_launch(void* const* d_in, const int* in_sizes, int n_in,
                              void* d_out, int out_size, void* d_ws, size_t ws_size,
                              hipStream_t stream)
{
  (void)in_sizes; (void)n_in; (void)out_size; (void)ws_size;

  const float* X    = (const float*)d_in[0];
  const float* cosT = (const float*)d_in[2];
  const float* sinT = (const float*)d_in[3];
  const float* Wq   = (const float*)d_in[4];
  const float* Wk   = (const float*)d_in[5];
  const float* Wv   = (const float*)d_in[6];
  const float* Wo   = (const float*)d_in[7];
  float* out = (float*)d_out;

  char* ws = (char*)d_ws;
  __bf16* Xbf   = (__bf16*)ws;  ws += (size_t)S_LEN * DIN       * sizeof(__bf16);
  __bf16* Qbf   = (__bf16*)ws;  ws += (size_t)S_LEN * NH  * DH  * sizeof(__bf16);
  __bf16* Kbf   = (__bf16*)ws;  ws += (size_t)S_LEN * NKV * DH  * sizeof(__bf16);
  __bf16* Vbf   = (__bf16*)ws;  ws += (size_t)S_LEN * NKV * DH  * sizeof(__bf16);
  __bf16* Ctxbf = (__bf16*)ws;

  const dim3 blk(256);

  // X -> bf16 (enables async A staging in all GEMMs)
  cvt_f32_bf16<<<(S_LEN * DIN / 4) / 256, 256, 0, stream>>>(X, Xbf);

  // QKV projections (bf16 outputs)
  gemm_bf16wmma<__bf16><<<dim3((NH  * DH) / 128, S_LEN / 128), blk, 0, stream>>>(
      Xbf, Wq, Qbf, S_LEN, NH * DH, DIN);
  gemm_bf16wmma<__bf16><<<dim3((NKV * DH) / 128, S_LEN / 128), blk, 0, stream>>>(
      Xbf, Wk, Kbf, S_LEN, NKV * DH, DIN);
  gemm_bf16wmma<__bf16><<<dim3((NKV * DH) / 128, S_LEN / 128), blk, 0, stream>>>(
      Xbf, Wv, Vbf, S_LEN, NKV * DH, DIN);

  // RoPE; fold 1/sqrt(DH) into Q
  const float qscale = 0.08838834764831845f;   // 1/sqrt(128)
  rope_kernel<<<(S_LEN * NH  * 64) / 256, 256, 0, stream>>>(Qbf, cosT, sinT, NH,  qscale);
  rope_kernel<<<(S_LEN * NKV * 64) / 256, 256, 0, stream>>>(Kbf, cosT, sinT, NKV, 1.0f);

  // causal GQA flash attention -> bf16 context
  gqa_flash_attn<<<dim3(S_LEN / 64, NH), dim3(128), 0, stream>>>(Qbf, Kbf, Vbf, Ctxbf);

  // output projection (fp32 out)
  gemm_bf16wmma<float><<<dim3(DIN / 128, S_LEN / 128), blk, 0, stream>>>(
      Ctxbf, Wo, out, S_LEN, DIN, DIN);
}
